// RoutedConvNeXtBlock_40407052321164
// MI455X (gfx1250) — matrix operand order repaired and checked by reading, not compiled, not run on hardware
//
#include <hip/hip_runtime.h>

// ---------------------------------------------------------------------------
// RoutedConvNeXtBlock for MI455X (gfx1250, wave32, WMMA 16x16x32 bf16)
// B=64 C=384 H=W=28 N=50176 HID=1536 E=3 K=2 R=8
// Round 2: M-tile 64 (4x B-fragment reuse), dynamic 140KB LDS,
//          async global->LDS staging (GLOBAL_LOAD_ASYNC_TO_LDS_B128).
// ---------------------------------------------------------------------------

typedef __attribute__((ext_vector_type(16))) __bf16 v16bf;
typedef __attribute__((ext_vector_type(8)))  float  v8f;

union FragBF { v16bf v; uint4 q[2]; };

__device__ inline v8f vzero() {
    v8f v = {0.f, 0.f, 0.f, 0.f, 0.f, 0.f, 0.f, 0.f};
    return v;
}

__device__ inline unsigned short f2bf(float f) {
    union { float f; unsigned int u; } c; c.f = f;
    unsigned int u = c.u;
    unsigned int r = (u + 0x7FFFu + ((u >> 16) & 1u)) >> 16;   // RNE
    return (unsigned short)r;
}

__device__ inline float bf2f(unsigned short h) {
    union { float f; unsigned int u; } c; c.u = ((unsigned int)h) << 16;
    return c.f;
}

__device__ inline float gelu_exact(float x) {
    return 0.5f * x * (1.0f + erff(x * 0.70710678118654752f));
}

// D = A(16x32 bf16) * B(32x16 bf16) + C(f32)
__device__ inline v8f wmma_bf16(v16bf a, v16bf b, v8f c) {
    return __builtin_amdgcn_wmma_f32_16x16x32_bf16(
        false, a, false, b, (short)0, c, false, false);
}

// ---- async global -> LDS (16B per lane), ASYNCcnt-tracked -----------------
__device__ inline void async_copy_b128(void* lds_dst, const void* gsrc) {
    unsigned loff = (unsigned)(unsigned long long)lds_dst; // addr[31:0] = LDS offset
    asm volatile("global_load_async_to_lds_b128 %0, %1, off"
                 :: "v"(loff), "v"(gsrc) : "memory");
}

__device__ inline void wait_async0() {
#if __has_builtin(__builtin_amdgcn_s_wait_asynccnt)
    __builtin_amdgcn_s_wait_asynccnt(0);
#else
    asm volatile("s_wait_asynccnt 0x0" ::: "memory");
#endif
}

// A fragment from LDS, row-major [M][K] (u16 units), row stride in u16.
// ISA 16-bit A 16x32 layout: lane<16 holds K{k0..k0+7, k0+16..k0+23},
// lane>=16 holds K{k0+8..k0+15, k0+24..k0+31}.
__device__ inline v16bf load_a(const unsigned short* base, int row0,
                               int stride_u16, int k0, int lane) {
    int m    = row0 + (lane & 15);
    int half = lane >> 4;
    const unsigned short* r = base + m * stride_u16;
    FragBF f;
    f.q[0] = *(const uint4*)(r + k0 + half * 8);
    f.q[1] = *(const uint4*)(r + k0 + 16 + half * 8);
    return f.v;
}

// B fragment from global, weights stored transposed [Ncol][K] bf16.
__device__ inline v16bf load_b(const unsigned short* W, int col0,
                               int kstride, int k0, int lane) {
    int col  = col0 + (lane & 15);
    int half = lane >> 4;
    const unsigned short* p = W + (size_t)col * kstride + k0 + half * 8;
    FragBF f;
    f.q[0] = *(const uint4*)(p);
    f.q[1] = *(const uint4*)(p + 16);
    return f.v;
}

// ---------------------------------------------------------------------------
// kW: convert+transpose fc1_w [384][1536] -> fc1T bf16 [1536][384]
//                       fc2_w [1536][384] -> fc2T bf16 [384][1536]
// ---------------------------------------------------------------------------
__global__ __launch_bounds__(256)
void kW_convert(const float* __restrict__ fc1w, const float* __restrict__ fc2w,
                unsigned short* __restrict__ fc1T, unsigned short* __restrict__ fc2T) {
    int i = blockIdx.x * 256 + threadIdx.x;
    const int Nw = 384 * 1536;
    if (i < Nw) {
        int k = i / 1536, f = i % 1536;
        fc1T[f * 384 + k] = f2bf(fc1w[i]);
    } else if (i < 2 * Nw) {
        int j = i - Nw;
        int k = j / 384, c = j % 384;
        fc2T[c * 1536 + k] = f2bf(fc2w[j]);
    }
}

// ---------------------------------------------------------------------------
// kA: one block per (b,h) image row: depthwise 7x7 conv -> per-pixel LN with
// per-sample fused affine -> y bf16 token-major [N][384]
// ---------------------------------------------------------------------------
__global__ __launch_bounds__(256)
void kA_conv_ln(const float* __restrict__ x, const float* __restrict__ convw,
                const float* __restrict__ convb,
                const float* __restrict__ ng, const float* __restrict__ nb,
                const float* __restrict__ domg, const float* __restrict__ domb,
                const float* __restrict__ rw, unsigned short* __restrict__ yws) {
    __shared__ float conv[384 * 28];
    __shared__ float geff[384];
    __shared__ float beff[384];

    const int bh = blockIdx.x;
    const int b  = bh / 28;
    const int h  = bh % 28;

    for (int c = threadIdx.x; c < 384; c += 256) {
        float g = ng[c], bb = nb[c];
#pragma unroll
        for (int e = 0; e < 3; e++) {
            float w = rw[b * 3 + e];
            g  += w * domg[e * 384 + c];
            bb += w * domb[e * 384 + c];
        }
        geff[c] = g;
        beff[c] = bb;
    }

    for (int idx = threadIdx.x; idx < 384 * 28; idx += 256) {
        int c = idx / 28, w = idx % 28;
        const float* xc = x + ((size_t)b * 384 + c) * 784;
        const float* wc = convw + c * 49;
        float acc = convb[c];
#pragma unroll
        for (int dy = -3; dy <= 3; dy++) {
            int hh = h + dy;
            if (hh < 0 || hh >= 28) continue;
#pragma unroll
            for (int dx = -3; dx <= 3; dx++) {
                int ww = w + dx;
                if (ww < 0 || ww >= 28) continue;
                acc += xc[hh * 28 + ww] * wc[(dy + 3) * 7 + (dx + 3)];
            }
        }
        conv[c * 28 + w] = acc;
    }
    __syncthreads();

    const int lane = threadIdx.x & 31;
    const int wv   = threadIdx.x >> 5;
    for (int p = wv; p < 28; p += 8) {
        float s1 = 0.f, s2 = 0.f;
        for (int c = lane; c < 384; c += 32) {
            float v = conv[c * 28 + p];
            s1 += v;
            s2 += v * v;
        }
#pragma unroll
        for (int m = 16; m >= 1; m >>= 1) {
            s1 += __shfl_xor(s1, m, 32);
            s2 += __shfl_xor(s2, m, 32);
        }
        float mu  = s1 * (1.f / 384.f);
        float var = s2 * (1.f / 384.f) - mu * mu;
        float rs  = rsqrtf(var + 1e-6f);
        size_t n  = ((size_t)b * 28 + h) * 28 + p;
        unsigned short* yo = yws + n * 384;
        for (int c = lane; c < 384; c += 32) {
            float v = (conv[c * 28 + p] - mu) * rs * geff[c] + beff[c];
            yo[c] = f2bf(v);
        }
    }
}

// ---------------------------------------------------------------------------
// kB: fused MLP + LoRA-MoE + layer scale + residual.  One block = 64 tokens.
// Dynamic LDS layout (143,616 B total):
//   [0,      50176)  y tile   : 64 rows x 392 u16 (49 uint4/row, padded)
//   [50176,  67584)  h chunk  : 64 rows x 136 u16 (17 uint4/row, padded)
//   [0,      99840)  OUT stage: 384 x 65 f32  (reuses y/h region at the end)
//   [99840, 105984)  gelu(down): 64 x 24 f32
//   [105984,106752)  comb     : 64 x 3 f32
//   [106752,143616)  w_up     : 3*8*384 f32 (LDS-resident for finalize)
// ---------------------------------------------------------------------------
#define LDSB_Y     0
#define LDSB_H     50176
#define LDSB_GD    99840
#define LDSB_COMB  105984
#define LDSB_WUP   106752
#define LDSB_TOTAL 143616

__global__ __launch_bounds__(256)
void kB_mlp(const unsigned short* __restrict__ yws,
            const unsigned short* __restrict__ fc1T,
            const unsigned short* __restrict__ fc2T,
            const float* __restrict__ fc1b, const float* __restrict__ fc2b,
            const float* __restrict__ wdown, const float* __restrict__ wup,
            const float* __restrict__ gamma,
            const float* __restrict__ tkp, const int* __restrict__ tki,
            const float* __restrict__ x, float* __restrict__ out) {
    extern __shared__ __align__(16) unsigned char smem[];
    unsigned short* Y    = (unsigned short*)(smem + LDSB_Y);
    unsigned short* Hc   = (unsigned short*)(smem + LDSB_H);
    float*          OUTS = (float*)(smem + LDSB_Y);      // overlaps Y/Hc
    float*          GD   = (float*)(smem + LDSB_GD);
    float*          COMB = (float*)(smem + LDSB_COMB);
    float*          WUP  = (float*)(smem + LDSB_WUP);

    const int tid  = threadIdx.x;
    const int lane = tid & 31;
    const int wv   = tid >> 5;
    const int half = lane >> 4;
    const int n0   = blockIdx.x * 64;

    // ---- async staging: y tile (3072 x b128) + w_up (2304 x b128) ----
    {
        uint4* dstY = (uint4*)Y;
        const uint4* srcY = (const uint4*)yws;
        for (int q = tid; q < 64 * 48; q += 256) {
            int r = q / 48, cc = q % 48;
            async_copy_b128(&dstY[r * 49 + cc], &srcY[(size_t)(n0 + r) * 48 + cc]);
        }
        uint4* dstW = (uint4*)WUP;
        const uint4* srcW = (const uint4*)wup;
        for (int q = tid; q < 2304; q += 256)
            async_copy_b128(&dstW[q], &srcW[q]);
    }

    // top-k combine weights per token (overlaps with async staging)
    if (tid < 64) {
        float c0 = 0.f, c1 = 0.f, c2 = 0.f;
#pragma unroll
        for (int k = 0; k < 2; k++) {
            int   e = tki[(n0 + tid) * 2 + k];
            float p = tkp[(n0 + tid) * 2 + k];
            if (e == 0) c0 += p; else if (e == 1) c1 += p; else c2 += p;
        }
        COMB[tid * 3 + 0] = c0;
        COMB[tid * 3 + 1] = c1;
        COMB[tid * 3 + 2] = c2;
    }

    wait_async0();
    __syncthreads();

    // ---- LoRA down-projection + exact gelu: 64 tokens x 24 (e,r) ----
    for (int j = tid; j < 64 * 24; j += 256) {
        int t = j & 63, er = j >> 6;
        int e = er >> 3, r = er & 7;
        const unsigned short* yr = Y + t * 392;
        const float* wd = wdown + (size_t)e * 384 * 8 + r;
        float s = 0.f;
        for (int c = 0; c < 384; c++) s += bf2f(yr[c]) * wd[c * 8];
        GD[t * 24 + er] = gelu_exact(s);
    }

    // fc2 accumulators: wave owns 3 channel-subtiles x 4 m-subtiles
    v8f acc[12];
#pragma unroll
    for (int i = 0; i < 12; i++) acc[i] = vzero();

    for (int kc = 0; kc < 12; kc++) {
        if (kc < 11)
            __builtin_prefetch(fc2T + (size_t)(wv * 48) * 1536 + (kc + 1) * 128, 0, 1);

        // ---- fc1: wave owns f-subtile f16 = wv; B frag reused over 4 m ----
        {
            const int fcol0 = kc * 128 + wv * 16;
            v8f a1[4];
#pragma unroll
            for (int i = 0; i < 4; i++) a1[i] = vzero();
#pragma unroll
            for (int k = 0; k < 384; k += 32) {
                v16bf bfr = load_b(fc1T, fcol0, 384, k, lane);
#pragma unroll
                for (int m16 = 0; m16 < 4; m16++) {
                    v16bf af = load_a(Y, m16 * 16, 392, k, lane);
                    a1[m16] = wmma_bf16(af, bfr, a1[m16]);
                }
            }
            float bias = fc1b[fcol0 + (lane & 15)];
            int colh = wv * 16 + (lane & 15);
#pragma unroll
            for (int m16 = 0; m16 < 4; m16++) {
#pragma unroll
                for (int r = 0; r < 8; r++) {
                    float hval = gelu_exact(a1[m16][r] + bias);
                    int trow = m16 * 16 + r + half * 8;
                    Hc[trow * 136 + colh] = f2bf(hval);
                }
            }
        }
        __syncthreads();

        // ---- fc2: accumulate 128-wide K slab; B frag reused over 4 m ----
#pragma unroll
        for (int kk = 0; kk < 128; kk += 32) {
            v16bf af[4];
#pragma unroll
            for (int m16 = 0; m16 < 4; m16++)
                af[m16] = load_a(Hc, m16 * 16, 136, kk, lane);
#pragma unroll
            for (int q = 0; q < 3; q++) {
                v16bf bfr = load_b(fc2T, (wv * 3 + q) * 16, 1536, kc * 128 + kk, lane);
#pragma unroll
                for (int m16 = 0; m16 < 4; m16++)
                    acc[q * 4 + m16] = wmma_bf16(af[m16], bfr, acc[q * 4 + m16]);
            }
        }
        __syncthreads();
    }

    // ---- stage accumulators channel-major [c][t] (stride 65) ----
#pragma unroll
    for (int q = 0; q < 3; q++) {
        int c = (wv * 3 + q) * 16 + (lane & 15);
#pragma unroll
        for (int m16 = 0; m16 < 4; m16++) {
#pragma unroll
            for (int r = 0; r < 8; r++) {
                int t = m16 * 16 + r + half * 8;
                OUTS[c * 65 + t] = acc[q * 4 + m16][r];
            }
        }
    }
    __syncthreads();

    // ---- finalize: + fc2 bias + LoRA up (SCALING==1); scale + residual ----
    for (int j = tid; j < 64 * 384; j += 256) {
        int t = j & 63, c = j >> 6;
        int n = n0 + t;
        int b = n / 784, hw = n % 784;
        float md = 0.f;
#pragma unroll
        for (int e = 0; e < 3; e++) {
            float we = COMB[t * 3 + e];
            float s = 0.f;
#pragma unroll
            for (int r = 0; r < 8; r++)
                s += GD[t * 24 + e * 8 + r] * WUP[(e * 8 + r) * 384 + c];
            md += we * s;
        }
        float val = OUTS[c * 65 + t] + fc2b[c] + md;
        size_t gi = ((size_t)b * 384 + c) * 784 + hw;
        out[gi] = gamma[c] * val + x[gi];
    }
}

// ---------------------------------------------------------------------------
extern "C" void kernel_launch(void* const* d_in, const int* in_sizes, int n_in,
                              void* d_out, int out_size, void* d_ws, size_t ws_size,
                              hipStream_t stream) {
    const float* x     = (const float*)d_in[0];
    const float* convw = (const float*)d_in[1];
    const float* convb = (const float*)d_in[2];
    const float* ng    = (const float*)d_in[3];
    const float* nb    = (const float*)d_in[4];
    const float* domg  = (const float*)d_in[5];
    const float* domb  = (const float*)d_in[6];
    const float* fc1w  = (const float*)d_in[7];
    const float* fc1b  = (const float*)d_in[8];
    const float* fc2w  = (const float*)d_in[9];
    const float* fc2b  = (const float*)d_in[10];
    const float* wdown = (const float*)d_in[11];
    const float* wup   = (const float*)d_in[12];
    const float* gamma = (const float*)d_in[13];
    const float* rw    = (const float*)d_in[14];
    const float* tkp   = (const float*)d_in[15];
    const int*   tki   = (const int*)d_in[16];
    float* out = (float*)d_out;

    char* ws = (char*)d_ws;
    unsigned short* fc1T = (unsigned short*)(ws);             // 1,179,648 B
    unsigned short* fc2T = (unsigned short*)(ws + 1179648);   // 1,179,648 B
    unsigned short* yws  = (unsigned short*)(ws + 2359296);   // 38,535,168 B

    hipLaunchKernelGGL(kW_convert, dim3(4608), dim3(256), 0, stream,
                       fc1w, fc2w, fc1T, fc2T);
    hipLaunchKernelGGL(kA_conv_ln, dim3(1792), dim3(256), 0, stream,
                       x, convw, convb, ng, nb, domg, domb, rw, yws);
    hipLaunchKernelGGL(kB_mlp, dim3(784), dim3(256), LDSB_TOTAL, stream,
                       yws, fc1T, fc2T, fc1b, fc2b, wdown, wup, gamma,
                       tkp, tki, x, out);
}
